// GATBert_32160715112913
// MI455X (gfx1250) — matrix operand order repaired and checked by reading, not compiled, not run on hardware
//
#include <hip/hip_runtime.h>
#include <cstdint>
#include <cstddef>

// ---------------- WMMA types (CDNA5 wave32) ----------------
typedef __attribute__((ext_vector_type(16))) __bf16 v16bf;
typedef __attribute__((ext_vector_type(8)))  float  v8f;

union AFrag { v16bf v; uint4 u[2]; };

__device__ __forceinline__ int   f2ord(float f){ int i = __float_as_int(f); return i < 0 ? (i ^ 0x7fffffff) : i; }
__device__ __forceinline__ float ord2f(int i){ int j = i < 0 ? (i ^ 0x7fffffff) : i; return __int_as_float(j); }

// ---------------- fills ----------------
__global__ void fill_f32_kernel(float* p, float v, long n){
  long i = (long)blockIdx.x * blockDim.x + threadIdx.x;
  if (i < n) p[i] = v;
}
__global__ void fill_i32_kernel(int* p, int v, long n){
  long i = (long)blockIdx.x * blockDim.x + threadIdx.x;
  if (i < n) p[i] = v;
}

// ---------------- bf16 conversions ----------------
// A: [M,K] f32 row-major -> [M,Kp] bf16 (zero-padded K)
__global__ void cvtA_kernel(const float* __restrict__ A, __bf16* __restrict__ O, long M, int K, int Kp){
  long i = (long)blockIdx.x * blockDim.x + threadIdx.x;
  if (i >= M * (long)Kp) return;
  long r = i / Kp; int k = (int)(i % Kp);
  O[i] = (k < K) ? (__bf16)A[r * (long)K + k] : (__bf16)0.0f;
}
// W: [K,N] f32 -> transposed padded [Nt,Kp] bf16
__global__ void cvtWT_kernel(const float* __restrict__ W, __bf16* __restrict__ O, int K, int N, int Kp, int Nt){
  long i = (long)blockIdx.x * blockDim.x + threadIdx.x;
  if (i >= (long)Nt * Kp) return;
  int k = (int)(i % Kp); int nrow = (int)(i / Kp);
  O[i] = (nrow < N && k < K) ? (__bf16)W[(size_t)k * N + nrow] : (__bf16)0.0f;
}

// ---------------- generic WMMA GEMM ----------------
// C[M,Nn] = A[M,Kp](bf16) * Bt[Nt16,Kp]^T (+bias) (+act).
// GATHER=1: A rows gathered fp32 (proteinsfile via target_id), converted in-registers.
// Block = 128 threads (4 waves). Wave computes 16 rows x 64 cols (4 accumulators).
// K-loop is software double-buffered: next step's fragments are loaded before the
// current step's WMMAs, so v_wmma overlaps with global_load latency.
template<int GATHER>
__global__ __launch_bounds__(128)
void gemm_wmma_kernel(const __bf16* __restrict__ Ab,
                      const float*  __restrict__ Agf, const int* __restrict__ gidx, int Lg, int ldaf,
                      const __bf16* __restrict__ Bt, int Kp, int Nt16,
                      const float* __restrict__ bias,
                      float* __restrict__ Cf, __bf16* __restrict__ Cb,
                      int M, int Nn, int act)
{
  const int lane = threadIdx.x & 31;
  const int wave = threadIdx.x >> 5;
  const int sel  = lane >> 4;        // 0: lanes 0-15, 1: lanes 16-31
  const int ml   = lane & 15;
  const int rowbase = blockIdx.x * 64 + wave * 16;
  const int colbase = blockIdx.y * 64;
  const int arow = rowbase + ml;

  const float*  afrow = nullptr;
  const __bf16* abrow = nullptr;
  if (GATHER) {
    int gr = gidx[arow / Lg] * Lg + (arow % Lg);
    afrow = Agf + (size_t)gr * ldaf + sel * 8;
  } else {
    abrow = Ab + (size_t)arow * Kp + sel * 8;
  }

  // B lane rows (clamped; rows of Bt are 64B aligned since Kp%32==0)
  const __bf16* bp[4];
  #pragma unroll
  for (int nt = 0; nt < 4; ++nt) {
    int c = colbase + nt * 16 + ml;
    c = c < Nt16 ? c : Nt16 - 1;
    bp[nt] = Bt + (size_t)c * Kp + sel * 8;
  }

  // A fragment: 16-bit A 16x32 layout -> lane holds K=[ka..ka+7] and [ka+16..ka+23], ka = k0 + sel*8
  auto loadA = [&](AFrag& a, int k0) {
    if (GATHER) {
      const float* ap = afrow + k0;
      float4 f0 = *reinterpret_cast<const float4*>(ap);
      float4 f1 = *reinterpret_cast<const float4*>(ap + 4);
      float4 f2 = *reinterpret_cast<const float4*>(ap + 16);
      float4 f3 = *reinterpret_cast<const float4*>(ap + 20);
      a.v[0]=(__bf16)f0.x;  a.v[1]=(__bf16)f0.y;  a.v[2]=(__bf16)f0.z;  a.v[3]=(__bf16)f0.w;
      a.v[4]=(__bf16)f1.x;  a.v[5]=(__bf16)f1.y;  a.v[6]=(__bf16)f1.z;  a.v[7]=(__bf16)f1.w;
      a.v[8]=(__bf16)f2.x;  a.v[9]=(__bf16)f2.y;  a.v[10]=(__bf16)f2.z; a.v[11]=(__bf16)f2.w;
      a.v[12]=(__bf16)f3.x; a.v[13]=(__bf16)f3.y; a.v[14]=(__bf16)f3.z; a.v[15]=(__bf16)f3.w;
    } else {
      const __bf16* ap = abrow + k0;
      a.u[0] = *reinterpret_cast<const uint4*>(ap);
      a.u[1] = *reinterpret_cast<const uint4*>(ap + 16);
    }
  };
  // B fragments: lane holds col n, K=[kb..kb+7] and [kb+16..kb+23], kb = k0 + sel*8
  auto loadB = [&](AFrag* b, int k0) {
    #pragma unroll
    for (int nt = 0; nt < 4; ++nt) {
      b[nt].u[0] = *reinterpret_cast<const uint4*>(bp[nt] + k0);
      b[nt].u[1] = *reinterpret_cast<const uint4*>(bp[nt] + k0 + 16);
    }
  };

  v8f acc[4] = {v8f{}, v8f{}, v8f{}, v8f{}};
  AFrag a, b[4];
  loadA(a, 0);
  loadB(b, 0);
  for (int k0 = 0; k0 < Kp; k0 += 32) {
    const int kn = (k0 + 32 < Kp) ? (k0 + 32) : k0;   // last iter: harmless reload
    AFrag an, bn[4];
    loadA(an, kn);          // prefetch next step while WMMAs below execute
    loadB(bn, kn);
    acc[0] = __builtin_amdgcn_wmma_f32_16x16x32_bf16(false, a.v, false, b[0].v, (short)0, acc[0], false, false);
    acc[1] = __builtin_amdgcn_wmma_f32_16x16x32_bf16(false, a.v, false, b[1].v, (short)0, acc[1], false, false);
    acc[2] = __builtin_amdgcn_wmma_f32_16x16x32_bf16(false, a.v, false, b[2].v, (short)0, acc[2], false, false);
    acc[3] = __builtin_amdgcn_wmma_f32_16x16x32_bf16(false, a.v, false, b[3].v, (short)0, acc[3], false, false);
    a = an;
    #pragma unroll
    for (int nt = 0; nt < 4; ++nt) b[nt] = bn[nt];
  }

  // epilogue: C/D layout -> lane col = ml, VGPR i row = i + 8*sel
  #pragma unroll
  for (int nt = 0; nt < 4; ++nt) {
    int n = colbase + nt * 16 + ml;
    if (n >= Nn) continue;
    float bv = bias ? bias[n] : 0.0f;
    #pragma unroll
    for (int i = 0; i < 8; ++i) {
      int r = rowbase + i + sel * 8;
      if (r >= M) continue;
      float v = acc[nt][i] + bv;
      if (act == 1) v = v > 0.0f ? v : 0.01f * v;      // LeakyReLU(0.01)
      else if (act == 2) v = tanhf(v);
      if (Cb) Cb[(size_t)r * Nn + n] = (__bf16)v;
      else    Cf[(size_t)r * Nn + n] = v;
    }
  }
}

// ---------------- GAT kernels ----------------
__global__ void att_reduce_kernel(const float* __restrict__ hW, const float* __restrict__ att_s,
                                  const float* __restrict__ att_d, float* __restrict__ as_,
                                  float* __restrict__ ad_, int n, int H, int C){
  long i = (long)blockIdx.x * blockDim.x + threadIdx.x;
  if (i >= (long)n * H) return;
  int node = (int)(i / H), h = (int)(i % H);
  const float* hp = hW + (size_t)node * H * C + (size_t)h * C;
  float ss = 0.f, sd = 0.f;
  for (int c = 0; c < C; ++c) { float v = hp[c]; ss += v * att_s[h * C + c]; sd += v * att_d[h * C + c]; }
  as_[i] = ss; ad_[i] = sd;
}

__global__ void edge_max_kernel(const float* __restrict__ as_, const float* __restrict__ ad_,
                                const int* __restrict__ ei, int* __restrict__ mI, int E, int n, int H){
  long i = (long)blockIdx.x * blockDim.x + threadIdx.x;
  long tot = (long)(E + n) * H;
  if (i >= tot) return;
  int e = (int)(i / H), h = (int)(i % H);
  int s, d;
  if (e < E) { s = ei[e]; d = ei[E + e]; } else { s = e - E; d = s; }
  float v = as_[(size_t)s * H + h] + ad_[(size_t)d * H + h];
  v = v > 0.f ? v : 0.2f * v;
  atomicMax(&mI[(size_t)d * H + h], f2ord(v));
}

__global__ void edge_exp_kernel(const float* __restrict__ as_, const float* __restrict__ ad_,
                                const int* __restrict__ ei, const int* __restrict__ mI,
                                float* __restrict__ den, float* __restrict__ ex_, int E, int n, int H){
  long i = (long)blockIdx.x * blockDim.x + threadIdx.x;
  long tot = (long)(E + n) * H;
  if (i >= tot) return;
  int e = (int)(i / H), h = (int)(i % H);
  int s, d;
  if (e < E) { s = ei[e]; d = ei[E + e]; } else { s = e - E; d = s; }
  float v = as_[(size_t)s * H + h] + ad_[(size_t)d * H + h];
  v = v > 0.f ? v : 0.2f * v;
  float m = ord2f(mI[(size_t)d * H + h]);
  float ex = expf(v - m);
  ex_[i] = ex;
  atomicAdd(&den[(size_t)d * H + h], ex);
}

__global__ void edge_accum_kernel(const float* __restrict__ hW, const float* __restrict__ ex_,
                                  const float* __restrict__ den, const int* __restrict__ ei,
                                  float* __restrict__ out, int E, int n, int H, int C){
  int e = blockIdx.x;
  int s, d;
  if (e < E) { s = ei[e]; d = ei[E + e]; } else { s = e - E; d = s; }
  int F = H * C;
  for (int f = threadIdx.x; f < F; f += blockDim.x) {
    int h = f / C;
    float alpha = ex_[(size_t)e * H + h] / (den[(size_t)d * H + h] + 1e-16f);
    atomicAdd(&out[(size_t)d * F + f], hW[(size_t)s * F + f] * alpha);
  }
}

__global__ void bias_act_kernel(float* __restrict__ x, const float* __restrict__ b, long tot, int F){
  long i = (long)blockIdx.x * blockDim.x + threadIdx.x;
  if (i >= tot) return;
  float v = x[i] + b[i % F];
  x[i] = v > 0.f ? v : 0.01f * v;
}

// ---------------- to_dense_batch ----------------
__global__ void count_kernel(const int* __restrict__ batch, int* __restrict__ counts, int n){
  int i = blockIdx.x * blockDim.x + threadIdx.x;
  if (i < n) atomicAdd(&counts[batch[i]], 1);
}
__global__ void offsets_kernel(const int* __restrict__ counts, int* __restrict__ offs, int n){
  if (threadIdx.x == 0 && blockIdx.x == 0) { int s = 0; for (int i = 0; i < n; ++i) { offs[i] = s; s += counts[i]; } }
}
__global__ void scatter_dense_kernel(const float* __restrict__ h, const int* __restrict__ batch,
                                     const int* __restrict__ offs, float* __restrict__ xd,
                                     int n, int F, int maxn){
  int i = blockIdx.x;
  int b = batch[i]; int pos = i - offs[b];
  if (pos < 0 || pos >= maxn) return;
  for (int f = threadIdx.x; f < F; f += blockDim.x)
    xd[((size_t)b * maxn + pos) * F + f] = h[(size_t)i * F + f];
}

// ---------------- LayerNorm + LeakyReLU, in place on bf16 [rows,512] ----------------
__global__ __launch_bounds__(128)
void ln_lrelu_kernel(__bf16* __restrict__ t, const float* __restrict__ g, const float* __restrict__ bb){
  __shared__ float red[128];
  size_t row = blockIdx.x;
  __bf16* p = t + row * 512;
  int tid = threadIdx.x;
  float v[4]; float s = 0.f;
  #pragma unroll
  for (int i = 0; i < 4; ++i) { v[i] = (float)p[tid + i * 128]; s += v[i]; }
  red[tid] = s; __syncthreads();
  for (int st = 64; st > 0; st >>= 1) { if (tid < st) red[tid] += red[tid + st]; __syncthreads(); }
  float mu = red[0] * (1.f / 512.f);
  __syncthreads();
  float q = 0.f;
  #pragma unroll
  for (int i = 0; i < 4; ++i) { float d = v[i] - mu; q += d * d; }
  red[tid] = q; __syncthreads();
  for (int st = 64; st > 0; st >>= 1) { if (tid < st) red[tid] += red[tid + st]; __syncthreads(); }
  float rs = rsqrtf(red[0] * (1.f / 512.f) + 1e-5f);
  #pragma unroll
  for (int i = 0; i < 4; ++i) {
    int idx = tid + i * 128;
    float y = (v[i] - mu) * rs * g[idx] + bb[idx];
    y = y > 0.f ? y : 0.01f * y;
    p[idx] = (__bf16)y;
  }
}

// ---------------- co-attention kernels (fp32) ----------------
// P[b,c,n] = sum_d W[c,d] * X[b,n,d]
__global__ void proj_kernel(const float* __restrict__ W, const float* __restrict__ X, float* __restrict__ P,
                            int Bn, int Cc, int Nn, int Dd){
  long i = (long)blockIdx.x * blockDim.x + threadIdx.x;
  if (i >= (long)Bn * Cc * Nn) return;
  int n = (int)(i % Nn); long r = i / Nn; int c = (int)(r % Cc); int b = (int)(r / Cc);
  const float* x = X + ((size_t)b * Nn + n) * Dd;
  const float* w = W + (size_t)c * Dd;
  float s = 0.f;
  for (int d = 0; d < Dd; ++d) s += w[d] * x[d];
  P[i] = s;
}
// Cm[b,l,n] = tanh(sum_d T[b,l,d] * WbX[b,d,n])
__global__ void cm_kernel(const float* __restrict__ T, const float* __restrict__ WbX, float* __restrict__ Cm,
                          int Bn, int L, int Nn, int D){
  long i = (long)blockIdx.x * blockDim.x + threadIdx.x;
  if (i >= (long)Bn * L * Nn) return;
  int n = (int)(i % Nn); long r = i / Nn; int l = (int)(r % L); int b = (int)(r / L);
  const float* tp = T + ((size_t)b * L + l) * D;
  const float* wp = WbX + (size_t)b * D * Nn + n;
  float s = 0.f;
  for (int d = 0; d < D; ++d) s += tp[d] * wp[(size_t)d * Nn];
  Cm[i] = tanhf(s);
}
// Hc[b,c,n] = tanh(WcX[b,c,n] + sum_l WpT[b,c,l]*Cm[b,l,n])
__global__ void hc_kernel(const float* __restrict__ WcX, const float* __restrict__ WpT,
                          const float* __restrict__ Cm, float* __restrict__ Hc,
                          int Bn, int Cc, int Nn, int L){
  long i = (long)blockIdx.x * blockDim.x + threadIdx.x;
  if (i >= (long)Bn * Cc * Nn) return;
  int n = (int)(i % Nn); long r = i / Nn; int c = (int)(r % Cc); int b = (int)(r / Cc);
  const float* wp = WpT + ((size_t)b * Cc + c) * L;
  const float* cm = Cm + (size_t)b * L * Nn + n;
  float s = WcX[i];
  for (int l = 0; l < L; ++l) s += wp[l] * cm[(size_t)l * Nn];
  Hc[i] = tanhf(s);
}
// Hp[b,c,l] = tanh(WpT[b,c,l] + sum_n WcX[b,c,n]*Cm[b,l,n])
__global__ void hp_kernel(const float* __restrict__ WpT, const float* __restrict__ WcX,
                          const float* __restrict__ Cm, float* __restrict__ Hp,
                          int Bn, int Cc, int L, int Nn){
  long i = (long)blockIdx.x * blockDim.x + threadIdx.x;
  if (i >= (long)Bn * Cc * L) return;
  int l = (int)(i % L); long r = i / L; int c = (int)(r % Cc); int b = (int)(r / Cc);
  const float* wc = WcX + ((size_t)b * Cc + c) * Nn;
  const float* cm = Cm + ((size_t)b * L + l) * Nn;
  float s = WpT[i];
  for (int n = 0; n < Nn; ++n) s += wc[n] * cm[n];
  Hp[i] = tanhf(s);
}
// sc[b,n] = sum_c w[c]*H[b,c,n]
__global__ void score_kernel(const float* __restrict__ w, const float* __restrict__ H, float* __restrict__ sc,
                             int Bn, int Cc, int Nn){
  long i = (long)blockIdx.x * blockDim.x + threadIdx.x;
  if (i >= (long)Bn * Nn) return;
  int n = (int)(i % Nn); int b = (int)(i / Nn);
  const float* h = H + (size_t)b * Cc * Nn + n;
  float s = 0.f;
  for (int c = 0; c < Cc; ++c) s += w[c] * h[(size_t)c * Nn];
  sc[i] = s;
}
__global__ __launch_bounds__(256)
void softmax_rows_kernel(float* __restrict__ x, int n){
  __shared__ float red[256];
  int r = blockIdx.x; int tid = threadIdx.x;
  float* row = x + (size_t)r * n;
  float m = -1e30f;
  for (int i = tid; i < n; i += 256) m = fmaxf(m, row[i]);
  red[tid] = m; __syncthreads();
  for (int s = 128; s > 0; s >>= 1) { if (tid < s) red[tid] = fmaxf(red[tid], red[tid + s]); __syncthreads(); }
  m = red[0]; __syncthreads();
  float ssum = 0.f;
  for (int i = tid; i < n; i += 256) { float e = expf(row[i] - m); row[i] = e; ssum += e; }
  red[tid] = ssum; __syncthreads();
  for (int s = 128; s > 0; s >>= 1) { if (tid < s) red[tid] += red[tid + s]; __syncthreads(); }
  float inv = 1.f / red[0];
  for (int i = tid; i < n; i += 256) row[i] *= inv;
}
// cp[b,0:128]=sum_n ac[b,n]*xd[b,n,:]; cp[b,128:256]=sum_l ap[b,l]*T[b,l,:]
__global__ void pool_kernel(const float* __restrict__ ac, const float* __restrict__ ap,
                            const float* __restrict__ xd, const float* __restrict__ T,
                            float* __restrict__ cp, int Bn, int Nn, int L, int D){
  long i = (long)blockIdx.x * blockDim.x + threadIdx.x;
  if (i >= (long)Bn * D) return;
  int d = (int)(i % D); int b = (int)(i / D);
  const float* a = ac + (size_t)b * Nn;
  const float* xp = xd + (size_t)b * Nn * D + d;
  float s = 0.f;
  for (int n = 0; n < Nn; ++n) s += a[n] * xp[(size_t)n * D];
  cp[(size_t)b * 2 * D + d] = s;
  const float* q = ap + (size_t)b * L;
  const float* tp = T + (size_t)b * L * D + d;
  float s2 = 0.f;
  for (int l = 0; l < L; ++l) s2 += q[l] * tp[(size_t)l * D];
  cp[(size_t)b * 2 * D + D + d] = s2;
}
__global__ void final_kernel(const float* __restrict__ c2o, const float* __restrict__ oW,
                             const float* __restrict__ ob, float* __restrict__ out, int Bn, int D){
  int b = blockIdx.x * blockDim.x + threadIdx.x;
  if (b >= Bn) return;
  float s = ob[0];
  const float* r = c2o + (size_t)b * D;
  for (int d = 0; d < D; ++d) s += r[d] * oW[d];
  out[b] = s;
}

// ---------------- host ----------------
static inline void launch_gemm(hipStream_t s, const __bf16* A, const float* Agf, const int* gidx, int Lg, int ldaf,
                               const __bf16* Bt, int Kp, int Nt16, const float* bias,
                               float* Cf, __bf16* Cb, int M, int Nn, int act){
  dim3 g((M + 63) / 64, (Nn + 63) / 64), b(128);
  if (gidx)
    gemm_wmma_kernel<1><<<g, b, 0, s>>>(A, Agf, gidx, Lg, ldaf, Bt, Kp, Nt16, bias, Cf, Cb, M, Nn, act);
  else
    gemm_wmma_kernel<0><<<g, b, 0, s>>>(A, Agf, gidx, Lg, ldaf, Bt, Kp, Nt16, bias, Cf, Cb, M, Nn, act);
}
#define G1D(n) dim3((unsigned)(((n) + 255) / 256)), dim3(256)

extern "C" void kernel_launch(void* const* d_in, const int* in_sizes, int n_in,
                              void* d_out, int out_size, void* d_ws, size_t ws_size,
                              hipStream_t stream) {
  (void)in_sizes; (void)n_in; (void)out_size; (void)ws_size;
  const float* x    = (const float*)d_in[0];
  const float* prot = (const float*)d_in[1];
  const float* g1W  = (const float*)d_in[2];
  const float* g1as = (const float*)d_in[3];
  const float* g1ad = (const float*)d_in[4];
  const float* g1b  = (const float*)d_in[5];
  const float* g2W  = (const float*)d_in[6];
  const float* g2as = (const float*)d_in[7];
  const float* g2ad = (const float*)d_in[8];
  const float* g2b  = (const float*)d_in[9];
  const float* fcW  = (const float*)d_in[10];
  const float* fcb  = (const float*)d_in[11];
  const float* ab1W = (const float*)d_in[12];
  const float* ab1b = (const float*)d_in[13];
  const float* lng  = (const float*)d_in[14];
  const float* lnb  = (const float*)d_in[15];
  const float* ab2W = (const float*)d_in[16];
  const float* ab2b = (const float*)d_in[17];
  const float* Wb   = (const float*)d_in[18];
  const float* Wc   = (const float*)d_in[19];
  const float* Wp   = (const float*)d_in[20];
  const float* whc  = (const float*)d_in[21];
  const float* whp  = (const float*)d_in[22];
  const float* c1W  = (const float*)d_in[23];
  const float* c1b  = (const float*)d_in[24];
  const float* c2W  = (const float*)d_in[25];
  const float* c2b  = (const float*)d_in[26];
  const float* oW   = (const float*)d_in[27];
  const float* ob   = (const float*)d_in[28];
  const int* ei     = (const int*)d_in[29];
  const int* batch  = (const int*)d_in[30];
  const int* tid    = (const int*)d_in[31];
  float* out        = (float*)d_out;
  char* ws          = (char*)d_ws;

  const int NN = 12800, EE = 51200, ETOT = 64000, Bc = 256, MAXN = 69, LSEQ = 512;
  const int H1 = 10, C1 = 78, Kp1 = 96, N1 = 780, Nt1 = 784;
  const int Kp2 = 800, N2 = 128;
  const int MX = Bc * MAXN;      // 17664
  const int ML = Bc * LSEQ;      // 131072
  const int ORD_NEG_INF = (int)0x807FFFFFu;  // f2ord(-inf)

  size_t off = 0;
  auto alloc = [&](size_t bytes)->size_t{ size_t o = off; off += (bytes + 255) & ~(size_t)255; return o; };
  #define WSP(T, o) reinterpret_cast<T*>(ws + (o))

  // persistent region
  size_t oW1t = alloc((size_t)Nt1 * Kp1 * 2);
  size_t oW2t = alloc((size_t)128 * Kp2 * 2);
  size_t oFcT = alloc((size_t)128 * 128 * 2);
  size_t oA1t = alloc((size_t)512 * 1280 * 2);
  size_t oA2t = alloc((size_t)128 * 512 * 2);
  size_t oC1t = alloc((size_t)1024 * 256 * 2);
  size_t oC2t = alloc((size_t)256 * 1024 * 2);
  size_t oXdo = alloc((size_t)MX * 128 * 4);
  size_t mark = off;

  // phase A (GAT + dense batch + fc1)
  size_t oXbf  = alloc((size_t)NN * Kp1 * 2);
  size_t oHW1  = alloc((size_t)NN * N1 * 4);
  size_t oAs1  = alloc((size_t)NN * H1 * 4);
  size_t oAd1  = alloc((size_t)NN * H1 * 4);
  size_t oM1   = alloc((size_t)NN * H1 * 4);
  size_t oDen1 = alloc((size_t)NN * H1 * 4);
  size_t oEx1  = alloc((size_t)ETOT * H1 * 4);
  size_t oG1o  = alloc((size_t)NN * N1 * 4);
  size_t oH1bf = alloc((size_t)NN * Kp2 * 2);
  size_t oHW2  = alloc((size_t)NN * N2 * 4);
  size_t oAs2  = alloc((size_t)NN * 4);
  size_t oAd2  = alloc((size_t)NN * 4);
  size_t oM2   = alloc((size_t)NN * 4);
  size_t oDen2 = alloc((size_t)NN * 4);
  size_t oEx2  = alloc((size_t)ETOT * 4);
  size_t oG2o  = alloc((size_t)NN * N2 * 4);
  size_t oCnt  = alloc((size_t)Bc * 4);
  size_t oOffB = alloc((size_t)Bc * 4);
  size_t oXdr  = alloc((size_t)MX * 128 * 4);
  size_t oXdb  = alloc((size_t)MX * 128 * 2);

  // phase B (protein + co-attention) reuses phase A region
  off = mark;
  size_t oT1  = alloc((size_t)ML * 512 * 2);
  size_t oT   = alloc((size_t)ML * 128 * 4);
  size_t oWbX = alloc((size_t)Bc * 128 * MAXN * 4);
  size_t oCm  = alloc((size_t)Bc * LSEQ * MAXN * 4);
  size_t oWcX = alloc((size_t)Bc * 64 * MAXN * 4);
  size_t oWpT = alloc((size_t)Bc * 64 * LSEQ * 4);
  size_t oHc  = alloc((size_t)Bc * 64 * MAXN * 4);
  size_t oHp  = alloc((size_t)Bc * 64 * LSEQ * 4);
  size_t oScC = alloc((size_t)Bc * MAXN * 4);
  size_t oScP = alloc((size_t)Bc * LSEQ * 4);
  size_t oCp  = alloc((size_t)Bc * 256 * 4);
  size_t oCpB = alloc((size_t)Bc * 256 * 2);
  size_t oC1o = alloc((size_t)Bc * 1024 * 4);
  size_t oC1b = alloc((size_t)Bc * 1024 * 2);
  size_t oC2o = alloc((size_t)Bc * 256 * 4);

  // ---- weight conversions (bf16 transposed, padded) ----
  cvtWT_kernel<<<G1D((long)Nt1 * Kp1), 0, stream>>>(g1W,  WSP(__bf16, oW1t), 78,   N1,  Kp1,  Nt1);
  cvtWT_kernel<<<G1D((long)128 * Kp2), 0, stream>>>(g2W,  WSP(__bf16, oW2t), N1,   128, Kp2,  128);
  cvtWT_kernel<<<G1D((long)128 * 128), 0, stream>>>(fcW,  WSP(__bf16, oFcT), 128,  128, 128,  128);
  cvtWT_kernel<<<G1D((long)512 * 1280), 0, stream>>>(ab1W, WSP(__bf16, oA1t), 1280, 512, 1280, 512);
  cvtWT_kernel<<<G1D((long)128 * 512), 0, stream>>>(ab2W, WSP(__bf16, oA2t), 512,  128, 512,  128);
  cvtWT_kernel<<<G1D((long)1024 * 256), 0, stream>>>(c1W,  WSP(__bf16, oC1t), 256,  1024, 256, 1024);
  cvtWT_kernel<<<G1D((long)256 * 1024), 0, stream>>>(c2W,  WSP(__bf16, oC2t), 1024, 256, 1024, 256);

  // ---- GAT layer 1 ----
  cvtA_kernel<<<G1D((long)NN * Kp1), 0, stream>>>(x, WSP(__bf16, oXbf), NN, 78, Kp1);
  launch_gemm(stream, WSP(__bf16, oXbf), nullptr, nullptr, 1, 0,
              WSP(__bf16, oW1t), Kp1, Nt1, nullptr, WSP(float, oHW1), nullptr, NN, N1, 0);
  att_reduce_kernel<<<G1D((long)NN * H1), 0, stream>>>(WSP(float, oHW1), g1as, g1ad,
              WSP(float, oAs1), WSP(float, oAd1), NN, H1, C1);
  fill_i32_kernel<<<G1D((long)NN * H1), 0, stream>>>(WSP(int, oM1), ORD_NEG_INF, (long)NN * H1);
  fill_f32_kernel<<<G1D((long)NN * H1), 0, stream>>>(WSP(float, oDen1), 0.f, (long)NN * H1);
  edge_max_kernel<<<G1D((long)ETOT * H1), 0, stream>>>(WSP(float, oAs1), WSP(float, oAd1), ei,
              WSP(int, oM1), EE, NN, H1);
  edge_exp_kernel<<<G1D((long)ETOT * H1), 0, stream>>>(WSP(float, oAs1), WSP(float, oAd1), ei,
              WSP(int, oM1), WSP(float, oDen1), WSP(float, oEx1), EE, NN, H1);
  fill_f32_kernel<<<G1D((long)NN * N1), 0, stream>>>(WSP(float, oG1o), 0.f, (long)NN * N1);
  edge_accum_kernel<<<dim3(ETOT), dim3(256), 0, stream>>>(WSP(float, oHW1), WSP(float, oEx1),
              WSP(float, oDen1), ei, WSP(float, oG1o), EE, NN, H1, C1);
  bias_act_kernel<<<G1D((long)NN * N1), 0, stream>>>(WSP(float, oG1o), g1b, (long)NN * N1, N1);

  // ---- GAT layer 2 ----
  cvtA_kernel<<<G1D((long)NN * Kp2), 0, stream>>>(WSP(float, oG1o), WSP(__bf16, oH1bf), NN, N1, Kp2);
  launch_gemm(stream, WSP(__bf16, oH1bf), nullptr, nullptr, 1, 0,
              WSP(__bf16, oW2t), Kp2, 128, nullptr, WSP(float, oHW2), nullptr, NN, N2, 0);
  att_reduce_kernel<<<G1D((long)NN), 0, stream>>>(WSP(float, oHW2), g2as, g2ad,
              WSP(float, oAs2), WSP(float, oAd2), NN, 1, 128);
  fill_i32_kernel<<<G1D((long)NN), 0, stream>>>(WSP(int, oM2), ORD_NEG_INF, (long)NN);
  fill_f32_kernel<<<G1D((long)NN), 0, stream>>>(WSP(float, oDen2), 0.f, (long)NN);
  edge_max_kernel<<<G1D((long)ETOT), 0, stream>>>(WSP(float, oAs2), WSP(float, oAd2), ei,
              WSP(int, oM2), EE, NN, 1);
  edge_exp_kernel<<<G1D((long)ETOT), 0, stream>>>(WSP(float, oAs2), WSP(float, oAd2), ei,
              WSP(int, oM2), WSP(float, oDen2), WSP(float, oEx2), EE, NN, 1);
  fill_f32_kernel<<<G1D((long)NN * N2), 0, stream>>>(WSP(float, oG2o), 0.f, (long)NN * N2);
  edge_accum_kernel<<<dim3(ETOT), dim3(256), 0, stream>>>(WSP(float, oHW2), WSP(float, oEx2),
              WSP(float, oDen2), ei, WSP(float, oG2o), EE, NN, 1, 128);
  bias_act_kernel<<<G1D((long)NN * N2), 0, stream>>>(WSP(float, oG2o), g2b, (long)NN * N2, N2);

  // ---- to_dense_batch + fc1 ----
  fill_i32_kernel<<<G1D((long)Bc), 0, stream>>>(WSP(int, oCnt), 0, (long)Bc);
  count_kernel<<<G1D((long)NN), 0, stream>>>(batch, WSP(int, oCnt), NN);
  offsets_kernel<<<dim3(1), dim3(1), 0, stream>>>(WSP(int, oCnt), WSP(int, oOffB), Bc);
  fill_f32_kernel<<<G1D((long)MX * 128), 0, stream>>>(WSP(float, oXdr), 0.f, (long)MX * 128);
  scatter_dense_kernel<<<dim3(NN), dim3(128), 0, stream>>>(WSP(float, oG2o), batch, WSP(int, oOffB),
              WSP(float, oXdr), NN, 128, MAXN);
  cvtA_kernel<<<G1D((long)MX * 128), 0, stream>>>(WSP(float, oXdr), WSP(__bf16, oXdb), MX, 128, 128);
  launch_gemm(stream, WSP(__bf16, oXdb), nullptr, nullptr, 1, 0,
              WSP(__bf16, oFcT), 128, 128, fcb, WSP(float, oXdo), nullptr, MX, 128, 1);

  // ---- protein branch: gather -> GEMM(1280->512) -> LN+lrelu -> GEMM(512->128)+lrelu ----
  launch_gemm(stream, nullptr, prot, tid, LSEQ, 1280,
              WSP(__bf16, oA1t), 1280, 512, ab1b, nullptr, WSP(__bf16, oT1), ML, 512, 0);
  ln_lrelu_kernel<<<dim3(ML), dim3(128), 0, stream>>>(WSP(__bf16, oT1), lng, lnb);
  launch_gemm(stream, WSP(__bf16, oT1), nullptr, nullptr, 1, 0,
              WSP(__bf16, oA2t), 512, 128, ab2b, WSP(float, oT), nullptr, ML, 128, 1);

  // ---- parallel co-attention ----
  proj_kernel<<<G1D((long)Bc * 128 * MAXN), 0, stream>>>(Wb, WSP(float, oXdo), WSP(float, oWbX), Bc, 128, MAXN, 128);
  cm_kernel<<<G1D((long)Bc * LSEQ * MAXN), 0, stream>>>(WSP(float, oT), WSP(float, oWbX), WSP(float, oCm),
              Bc, LSEQ, MAXN, 128);
  proj_kernel<<<G1D((long)Bc * 64 * MAXN), 0, stream>>>(Wc, WSP(float, oXdo), WSP(float, oWcX), Bc, 64, MAXN, 128);
  proj_kernel<<<G1D((long)Bc * 64 * LSEQ), 0, stream>>>(Wp, WSP(float, oT), WSP(float, oWpT), Bc, 64, LSEQ, 128);
  hc_kernel<<<G1D((long)Bc * 64 * MAXN), 0, stream>>>(WSP(float, oWcX), WSP(float, oWpT), WSP(float, oCm),
              WSP(float, oHc), Bc, 64, MAXN, LSEQ);
  hp_kernel<<<G1D((long)Bc * 64 * LSEQ), 0, stream>>>(WSP(float, oWpT), WSP(float, oWcX), WSP(float, oCm),
              WSP(float, oHp), Bc, 64, LSEQ, MAXN);
  score_kernel<<<G1D((long)Bc * MAXN), 0, stream>>>(whc, WSP(float, oHc), WSP(float, oScC), Bc, 64, MAXN);
  score_kernel<<<G1D((long)Bc * LSEQ), 0, stream>>>(whp, WSP(float, oHp), WSP(float, oScP), Bc, 64, LSEQ);
  softmax_rows_kernel<<<dim3(Bc), dim3(256), 0, stream>>>(WSP(float, oScC), MAXN);
  softmax_rows_kernel<<<dim3(Bc), dim3(256), 0, stream>>>(WSP(float, oScP), LSEQ);
  pool_kernel<<<G1D((long)Bc * 128), 0, stream>>>(WSP(float, oScC), WSP(float, oScP),
              WSP(float, oXdo), WSP(float, oT), WSP(float, oCp), Bc, MAXN, LSEQ, 128);

  // ---- head MLP ----
  cvtA_kernel<<<G1D((long)Bc * 256), 0, stream>>>(WSP(float, oCp), WSP(__bf16, oCpB), Bc, 256, 256);
  launch_gemm(stream, WSP(__bf16, oCpB), nullptr, nullptr, 1, 0,
              WSP(__bf16, oC1t), 256, 1024, c1b, WSP(float, oC1o), nullptr, Bc, 1024, 1);
  cvtA_kernel<<<G1D((long)Bc * 1024), 0, stream>>>(WSP(float, oC1o), WSP(__bf16, oC1b), Bc, 1024, 1024);
  launch_gemm(stream, WSP(__bf16, oC1b), nullptr, nullptr, 1, 0,
              WSP(__bf16, oC2t), 1024, 256, c2b, WSP(float, oC2o), nullptr, Bc, 256, 1);
  final_kernel<<<dim3(1), dim3(256), 0, stream>>>(WSP(float, oC2o), oW, ob, out, Bc, 256);
}
